// AttentionFlow_39195871543771
// MI455X (gfx1250) — matrix operand order repaired and checked by reading, not compiled, not run on hardware
//
#include <hip/hip_runtime.h>
#include <hip/hip_bf16.h>

// ---------------- problem constants ----------------
#define BB 8
#define CC 2048
#define QQ 128
#define DD 256
#define FD 1024   // 4*D

typedef __attribute__((ext_vector_type(16))) __bf16 v16bf;
typedef __attribute__((ext_vector_type(8)))  float  v8f;

__device__ __forceinline__ v8f wmma_bf16(v16bf a, v16bf b, v8f c) {
  // D(f32 16x16) = A(bf16 16x32) * B(bf16 32x16) + C
  return __builtin_amdgcn_wmma_f32_16x16x32_bf16(false, a, false, b, (short)0, c, false, false);
}

// low 32 bits of a flat shared-memory address == wave-relative LDS byte offset
__device__ __forceinline__ unsigned lds_off_of(const void* p) {
  return (unsigned)(unsigned long long)(uintptr_t)p;
}
// CDNA5 async DMA: copy 16B global -> LDS, tracked with ASYNCcnt (ISA 08 §4)
__device__ __forceinline__ void async_copy_b128(unsigned ldsOff, const void* g) {
  asm volatile("global_load_async_to_lds_b128 %0, %1, off"
               :: "v"(ldsOff), "v"(g) : "memory");
}
__device__ __forceinline__ void wait_async0() {
  asm volatile("s_wait_asynccnt 0" ::: "memory");
}

// ---------------- K1a: per-row dot with weight vector (sc, sq) ----------------
__global__ __launch_bounds__(256) void k_rowdot(const float* __restrict__ src,
                                                const float* __restrict__ w,
                                                float* __restrict__ out, int nrows) {
  int gwave = (blockIdx.x * 256 + threadIdx.x) >> 5;
  int lane  = threadIdx.x & 31;
  if (gwave >= nrows) return;
  const float* r = src + (size_t)gwave * DD;
  float s = 0.f;
  #pragma unroll
  for (int d = lane; d < DD; d += 32) s += r[d] * w[d];
  #pragma unroll
  for (int m = 16; m >= 1; m >>= 1) s += __shfl_xor(s, m, 32);
  if (lane == 0) out[gwave] = s;
}

// ---------------- K1b: questions -> bf16 (row-major + transposed) ----------------
__global__ __launch_bounds__(256) void k_prep_q(const float* __restrict__ q,
                                                __bf16* __restrict__ qA,
                                                __bf16* __restrict__ qT) {
  int idx = blockIdx.x * 256 + threadIdx.x;           // B*Q*D = 262144
  if (idx >= BB * QQ * DD) return;
  int d  = idx & (DD - 1);
  int qq = (idx >> 8) & (QQ - 1);
  int b  = idx >> 15;
  __bf16 v = (__bf16)q[idx];
  qA[idx] = v;                                        // [b][q][d]
  qT[((size_t)b * DD + d) * QQ + qq] = v;             // [b][d][q]
}

// ---------------- K1c: f32 -> bf16 bulk convert (W2, contexts) ----------------
__global__ __launch_bounds__(256) void k_prep_bf16(const float* __restrict__ src,
                                                   __bf16* __restrict__ dst, int n) {
  int idx = blockIdx.x * 256 + threadIdx.x;
  if (idx < n) dst[idx] = (__bf16)src[idx];
}

// ---------------- K2: sim -> softmax_q -> u_tilde (+ sim_max) ----------------
// grid = B * (C/128); 256 threads = 8 waves; wave owns 16 context rows.
__global__ __launch_bounds__(256) void k_sim_softmax_u(
    const __bf16* __restrict__ ctxbf, const float* __restrict__ wsim,
    const float* __restrict__ scv, const float* __restrict__ sqv,
    const unsigned char* __restrict__ qmask,
    const __bf16* __restrict__ qA, const __bf16* __restrict__ qT,
    float* __restrict__ smax, __bf16* __restrict__ ubf) {
  __shared__ __bf16 aLds[8][16 * QQ];                 // 32 KB: per-wave prob tiles

  const int tid  = threadIdx.x;
  const int wave = tid >> 5;
  const int lane = tid & 31;
  const int nloc = lane & 15;
  const int hf   = lane >> 4;
  const int b    = blockIdx.x >> 4;                   // 16 ctiles per batch
  const int ct   = blockIdx.x & 15;
  const int row0 = ct * 128 + wave * 16;
  const float* wcq = wsim + 2 * DD;

  // ---- A fragments for sim: (ctx * wcq), 16x256 bf16, 8 K-steps of 32
  // A layout (16-bit, 16x32): lane half hf holds K chunks [hf*8, hf*8+8) and [16+hf*8, ...)
  v16bf afr[8];
  {
    const __bf16* crow = ctxbf + ((size_t)b * CC + row0 + nloc) * DD;
    #pragma unroll
    for (int ks = 0; ks < 8; ++ks) {
      int kb = ks * 32 + hf * 8;
      #pragma unroll
      for (int i = 0; i < 8; ++i) {
        afr[ks][i]     = (__bf16)((float)crow[kb + i]      * wcq[kb + i]);
        afr[ks][i + 8] = (__bf16)((float)crow[kb + 16 + i] * wcq[kb + 16 + i]);
      }
    }
  }

  // ---- sim = A @ questions^T : 8 N-tiles of 16 q-columns
  v8f acc[8];
  #pragma unroll
  for (int nt = 0; nt < 8; ++nt) {
    v8f c = {0.f, 0.f, 0.f, 0.f, 0.f, 0.f, 0.f, 0.f};
    // B column n (= question q) is a contiguous row of qA
    const __bf16* qrow = qA + ((size_t)b * QQ + nt * 16 + nloc) * DD;
    #pragma unroll
    for (int ks = 0; ks < 8; ++ks) {
      v16bf bf;
      #pragma unroll
      for (int i = 0; i < 16; ++i) bf[i] = qrow[ks * 32 + hf * 8 + (i & 7) + ((i >> 3) << 4)];
      c = wmma_bf16(afr[ks], bf, c);
    }
    acc[nt] = c;
  }

  // ---- add sc + sq, mask, row softmax over q (rows = row0 + hf*8 + r)
  float scr[8], rmax[8], rsum[8];
  #pragma unroll
  for (int r = 0; r < 8; ++r) {
    scr[r]  = scv[b * CC + row0 + hf * 8 + r];
    rmax[r] = -3.0e38f;
    rsum[r] = 0.f;
  }
  #pragma unroll
  for (int nt = 0; nt < 8; ++nt) {
    int q = nt * 16 + nloc;
    float sqc = sqv[b * QQ + q];
    bool mk = qmask[b * QQ + q] != 0;
    #pragma unroll
    for (int r = 0; r < 8; ++r) {
      float v = mk ? (acc[nt][r] + scr[r] + sqc) : -1.0e30f;
      acc[nt][r] = v;
      rmax[r] = fmaxf(rmax[r], v);
    }
  }
  #pragma unroll
  for (int r = 0; r < 8; ++r) {
    #pragma unroll
    for (int m = 1; m <= 8; m <<= 1)                  // reduce within 16-lane half
      rmax[r] = fmaxf(rmax[r], __shfl_xor(rmax[r], m, 32));
  }
  #pragma unroll
  for (int nt = 0; nt < 8; ++nt) {
    #pragma unroll
    for (int r = 0; r < 8; ++r) {
      float e = __expf(acc[nt][r] - rmax[r]);
      acc[nt][r] = e;
      rsum[r] += e;
    }
  }
  #pragma unroll
  for (int r = 0; r < 8; ++r) {
    #pragma unroll
    for (int m = 1; m <= 8; m <<= 1)
      rsum[r] += __shfl_xor(rsum[r], m, 32);
  }
  if (nloc == 0) {
    #pragma unroll
    for (int r = 0; r < 8; ++r) smax[b * CC + row0 + hf * 8 + r] = rmax[r];
  }

  // ---- probabilities to LDS (D-layout -> row-major), then reload as A fragments
  #pragma unroll
  for (int nt = 0; nt < 8; ++nt) {
    #pragma unroll
    for (int r = 0; r < 8; ++r)
      aLds[wave][(hf * 8 + r) * QQ + nt * 16 + nloc] = (__bf16)(acc[nt][r] / rsum[r]);
  }
  __syncthreads();

  // ---- u_tilde = a(16x128) @ questions(128x256) : 4 K-steps, 16 N-tiles over D
  v16bf ufr[4];
  #pragma unroll
  for (int ks = 0; ks < 4; ++ks) {
    int kb = ks * 32 + hf * 8;
    #pragma unroll
    for (int i = 0; i < 8; ++i) {
      ufr[ks][i]     = aLds[wave][nloc * QQ + kb + i];
      ufr[ks][i + 8] = aLds[wave][nloc * QQ + kb + 16 + i];
    }
  }
  #pragma unroll
  for (int dt = 0; dt < 16; ++dt) {
    v8f c = {0.f, 0.f, 0.f, 0.f, 0.f, 0.f, 0.f, 0.f};
    // B column n (= feature d) is a contiguous row of qT
    const __bf16* qcol = qT + ((size_t)b * DD + dt * 16 + nloc) * QQ;
    #pragma unroll
    for (int ks = 0; ks < 4; ++ks) {
      v16bf bf;
      #pragma unroll
      for (int i = 0; i < 16; ++i) bf[i] = qcol[ks * 32 + hf * 8 + (i & 7) + ((i >> 3) << 4)];
      c = wmma_bf16(ufr[ks], bf, c);
    }
    #pragma unroll
    for (int r = 0; r < 8; ++r)
      ubf[((size_t)b * CC + row0 + hf * 8 + r) * DD + dt * 16 + nloc] = (__bf16)c[r];
  }
}

// ---------------- K3: bw softmax over c + h_tilde ----------------
__global__ __launch_bounds__(256) void k_bw_h(const float* __restrict__ smax,
                                              const unsigned char* __restrict__ cmask,
                                              const float* __restrict__ ctx,
                                              float* __restrict__ hvec) {
  __shared__ float bwL[CC];                           // 8 KB
  __shared__ float red[256];
  int b = blockIdx.x, tid = threadIdx.x;
  float lm = -3.0e38f;
  for (int c = tid; c < CC; c += 256) {
    float v = cmask[b * CC + c] ? smax[b * CC + c] : -1.0e30f;
    bwL[c] = v;
    lm = fmaxf(lm, v);
  }
  red[tid] = lm;
  __syncthreads();
  for (int s = 128; s > 0; s >>= 1) {
    if (tid < s) red[tid] = fmaxf(red[tid], red[tid + s]);
    __syncthreads();
  }
  float mx = red[0];
  __syncthreads();
  float ls = 0.f;
  for (int c = tid; c < CC; c += 256) {
    float e = __expf(bwL[c] - mx);
    bwL[c] = e;
    ls += e;
  }
  red[tid] = ls;
  __syncthreads();
  for (int s = 128; s > 0; s >>= 1) {
    if (tid < s) red[tid] += red[tid + s];
    __syncthreads();
  }
  float inv = 1.0f / red[0];
  __syncthreads();
  float acc = 0.f;                                    // thread tid owns feature d=tid
  for (int c = 0; c < CC; ++c) acc += bwL[c] * ctx[((size_t)b * CC + c) * DD + tid];
  hvec[b * DD + tid] = acc * inv;
}

// ---------------- K4: g = mega @ W2^T + b2 ----------------
// grid = B * (C/64) * 2 ; 256 threads = 8 waves; LDS holds 64x1024 bf16 mega tile
// (128 KB -- gfx1250 WGP has 320 KB). Each wave owns 64 output columns and ALL
// four 16-row M-subtiles, so every W2 fragment is fetched once per workgroup and
// feeds 4 WMMAs. Quarters 0 (ctx) / 1 (u) of mega arrive via async global->LDS
// DMA; quarters 2 (u*ctx) / 3 (h*ctx) are computed in-place from the LDS data.
__global__ __launch_bounds__(256) void k_gemm(const __bf16* __restrict__ ctxbf,
                                              const __bf16* __restrict__ ubf,
                                              const float* __restrict__ hvec,
                                              const __bf16* __restrict__ w2bf,
                                              const float* __restrict__ bias2,
                                              float* __restrict__ g) {
  __shared__ __bf16 megaLds[64 * FD];                 // 128 KB
  const int tid = threadIdx.x;
  const int wg = blockIdx.x;
  const int nhalf = wg & 1;
  const int ct = (wg >> 1) & 31;
  const int b  = wg >> 6;
  const int c0 = ct * 64;

  // ---- async-stage quarters 0/1: 64 rows x 2 x 512B = 4096 x 16B chunks
  for (int ch = tid; ch < 4096; ch += 256) {
    int row  = ch >> 6;                               // 64 chunks per row
    int c2   = ch & 63;
    int qsel = c2 >> 5;                               // 0 -> ctx, 1 -> u
    int part = c2 & 31;                               // 16B chunk within 512B
    size_t cix = (size_t)b * CC + c0 + row;
    const __bf16* src = (qsel == 0 ? ctxbf : ubf) + cix * DD + part * 8;
    unsigned dst = lds_off_of(&megaLds[row * FD + qsel * 256 + part * 8]);
    async_copy_b128(dst, src);
  }
  wait_async0();
  __syncthreads();

  // ---- compute quarters 2/3 from LDS-resident bf16 (no extra global traffic)
  for (int e = tid; e < 64 * 256; e += 256) {
    int row = e >> 8, d = e & 255;
    float cb = (float)megaLds[row * FD + d];
    float ub = (float)megaLds[row * FD + 256 + d];
    megaLds[row * FD + 512 + d] = (__bf16)(ub * cb);
    megaLds[row * FD + 768 + d] = (__bf16)(hvec[b * DD + d] * cb);
  }
  __syncthreads();

  const int wave = tid >> 5, lane = tid & 31;
  const int nloc = lane & 15, hf = lane >> 4;
  const int ncol = nhalf * 512 + wave * 64;           // 64 output cols per wave

  v8f acc[4][4];                                      // [mtile][ntile]
  #pragma unroll
  for (int mt = 0; mt < 4; ++mt)
    #pragma unroll
    for (int nt = 0; nt < 4; ++nt)
      acc[mt][nt] = (v8f){0.f, 0.f, 0.f, 0.f, 0.f, 0.f, 0.f, 0.f};

  for (int kk = 0; kk < FD; kk += 32) {
    // A fragments for the four 16-row subtiles (from LDS)
    v16bf afr[4];
    #pragma unroll
    for (int mt = 0; mt < 4; ++mt) {
      int base = (mt * 16 + nloc) * FD + kk + hf * 8;
      #pragma unroll
      for (int i = 0; i < 8; ++i) {
        afr[mt][i]     = megaLds[base + i];
        afr[mt][i + 8] = megaLds[base + 16 + i];
      }
    }
    // warm WGP$ for the next K-slice of W2 (L2-resident, 2 MB)
    if (kk + 32 < FD)
      __builtin_prefetch(w2bf + (size_t)(ncol + nloc) * FD + kk + 32 + hf * 16, 0, 3);
    #pragma unroll
    for (int nt = 0; nt < 4; ++nt) {
      // B column j = row j of W2 -> contiguous over k; fetched once, used 4x
      const __bf16* wrow = w2bf + (size_t)(ncol + nt * 16 + nloc) * FD + kk + hf * 8;
      v16bf bf;
      #pragma unroll
      for (int i = 0; i < 16; ++i) bf[i] = wrow[(i & 7) + ((i >> 3) << 4)];
      #pragma unroll
      for (int mt = 0; mt < 4; ++mt)
        acc[mt][nt] = wmma_bf16(afr[mt], bf, acc[mt][nt]);
    }
  }

  #pragma unroll
  for (int nt = 0; nt < 4; ++nt) {
    int j = ncol + nt * 16 + nloc;
    float bv = bias2[j];
    #pragma unroll
    for (int mt = 0; mt < 4; ++mt) {
      #pragma unroll
      for (int r = 0; r < 8; ++r)
        g[((size_t)b * CC + c0 + mt * 16 + hf * 8 + r) * FD + j] = acc[mt][nt][r] + bv;
    }
  }
}

// ---------------- host-side launch ----------------
extern "C" void kernel_launch(void* const* d_in, const int* in_sizes, int n_in,
                              void* d_out, int out_size, void* d_ws, size_t ws_size,
                              hipStream_t stream) {
  const float* questions = (const float*)d_in[0];            // B,Q,D
  const float* contexts  = (const float*)d_in[1];            // B,C,D
  const unsigned char* qmask = (const unsigned char*)d_in[2];// B,Q (bool)
  const unsigned char* cmask = (const unsigned char*)d_in[3];// B,C (bool)
  const float* wsim = (const float*)d_in[4];                 // 3D
  const float* W2   = (const float*)d_in[5];                 // 4D,4D
  const float* b2   = (const float*)d_in[6];                 // 4D
  float* g = (float*)d_out;

  // workspace layout (~19.5 MB)
  char* ws = (char*)d_ws;
  float* sqv  = (float*)ws;                 ws += (size_t)BB * QQ * sizeof(float);
  float* scv  = (float*)ws;                 ws += (size_t)BB * CC * sizeof(float);
  float* smax = (float*)ws;                 ws += (size_t)BB * CC * sizeof(float);
  float* hvec = (float*)ws;                 ws += (size_t)BB * DD * sizeof(float);
  __bf16* qA    = (__bf16*)ws;              ws += (size_t)BB * QQ * DD * sizeof(__bf16);
  __bf16* qT    = (__bf16*)ws;              ws += (size_t)BB * DD * QQ * sizeof(__bf16);
  __bf16* w2bf  = (__bf16*)ws;              ws += (size_t)FD * FD * sizeof(__bf16);
  __bf16* ubf   = (__bf16*)ws;              ws += (size_t)BB * CC * DD * sizeof(__bf16);
  __bf16* ctxbf = (__bf16*)ws;              ws += (size_t)BB * CC * DD * sizeof(__bf16);

  k_rowdot<<<(BB * CC + 7) / 8, 256, 0, stream>>>(contexts, wsim, scv, BB * CC);
  k_rowdot<<<(BB * QQ + 7) / 8, 256, 0, stream>>>(questions, wsim + DD, sqv, BB * QQ);
  k_prep_q<<<(BB * QQ * DD + 255) / 256, 256, 0, stream>>>(questions, qA, qT);
  k_prep_bf16<<<(FD * FD + 255) / 256, 256, 0, stream>>>(W2, w2bf, FD * FD);
  k_prep_bf16<<<(BB * CC * DD + 255) / 256, 256, 0, stream>>>(contexts, ctxbf, BB * CC * DD);
  k_sim_softmax_u<<<BB * (CC / 128), 256, 0, stream>>>(ctxbf, wsim, scv, sqv, qmask,
                                                       qA, qT, smax, ubf);
  k_bw_h<<<BB, 256, 0, stream>>>(smax, cmask, contexts, hvec);
  k_gemm<<<BB * (CC / 64) * 2, 256, 0, stream>>>(ctxbf, ubf, hvec, w2bf, b2, g);
}
